// T5Gemma2MergedAttention_61675730371278
// MI455X (gfx1250) — matrix operand order repaired and checked by reading, not compiled, not run on hardware
//
#include <hip/hip_runtime.h>
#include <cstdint>
#include <cstddef>

// ---------------- types ----------------
typedef __attribute__((ext_vector_type(16))) __bf16 v16bf;
typedef __attribute__((ext_vector_type(8)))  __bf16 v8bf;
typedef __attribute__((ext_vector_type(4)))  __bf16 v4bf;
typedef __attribute__((ext_vector_type(8)))  float  v8f;
typedef __attribute__((ext_vector_type(4)))  float  v4f;
typedef __attribute__((ext_vector_type(4)))  unsigned int v4u;
typedef __attribute__((ext_vector_type(8)))  int v8i;
typedef __attribute__((ext_vector_type(4)))  int v4i;

// ---------------- problem sizes ----------------
// B=2, S=2048, ENC=2048, HID=2048, HQ=8, HKV=4, D=256, L=S+ENC=4096
#define CB   2
#define CS   2048
#define CENC 2048
#define CHID 2048
#define CHQ  8
#define CHKV 4
#define CD   256
#define CL   4096
#define SCALING 0.0625f     // 256^-0.5
#define SOFTCAP 50.0f
#define NEGINF (-1.0e9f)

// ---------------- workspace layout (bytes) ----------------
static constexpr size_t OFF_HIDBF  = 0;                       // 4096*2048 bf16
static constexpr size_t OFF_ENCBF  = OFF_HIDBF  + 16777216;
static constexpr size_t OFF_WQBF   = OFF_ENCBF  + 16777216;   // 2048*2048 bf16
static constexpr size_t OFF_WKBF   = OFF_WQBF   + 8388608;    // 1024*2048 bf16
static constexpr size_t OFF_WVBF   = OFF_WKBF   + 4194304;
static constexpr size_t OFF_WOBF   = OFF_WVBF   + 4194304;    // 2048*2048 bf16
static constexpr size_t OFF_QRAW   = OFF_WOBF   + 8388608;    // 4096*2048 f32
static constexpr size_t OFF_KSELF  = OFF_QRAW   + 33554432;   // 4096*1024 f32
static constexpr size_t OFF_VSELF  = OFF_KSELF  + 16777216;
static constexpr size_t OFF_KCROSS = OFF_VSELF  + 16777216;
static constexpr size_t OFF_VCROSS = OFF_KCROSS + 16777216;
static constexpr size_t OFF_QN     = OFF_VCROSS + 16777216;   // [B,HQ,S,D] bf16
static constexpr size_t OFF_KBF    = OFF_QN     + 16777216;   // [B,HKV,L,D] bf16
static constexpr size_t OFF_VBF    = OFF_KBF    + 16777216;   // [B,HKV,L,D] bf16
static constexpr size_t OFF_ATTN   = OFF_VBF    + 16777216;   // [B*S, HQ*D] bf16

// ---------------- fp32 -> bf16 (vectorized: 16B load, 8B store per thread) ----------------
__global__ void k_f2bf(const float* __restrict__ in, __bf16* __restrict__ out, int n) {
    int i = (blockIdx.x * blockDim.x + threadIdx.x) * 4;
    if (i < n) {
        v4f v = *(const v4f*)(in + i);
        v4bf o;
        o[0] = (__bf16)v[0]; o[1] = (__bf16)v[1];
        o[2] = (__bf16)v[2]; o[3] = (__bf16)v[3];
        *(v4bf*)(out + i) = o;
    }
}

// ---------------- TDM 2D tile load (bf16 elements) ----------------
__device__ __forceinline__ void tdm_load_2d(unsigned lds_off, const void* gptr,
                                            unsigned tile_d0, unsigned tile_d1,
                                            unsigned tdim0, unsigned tdim1,
                                            unsigned long long stride0) {
    unsigned long long ga = (unsigned long long)gptr;
    v4u g0;
    g0[0] = 1u;                                  // count=1, user mode, no gather
    g0[1] = lds_off;                             // LDS byte address
    g0[2] = (unsigned)(ga & 0xFFFFFFFFull);      // global_addr[31:0]
    g0[3] = (unsigned)((ga >> 32) & 0x1FFFFFFull) | (2u << 30);  // addr[56:32] | type=2
    v8i g1;
    g1[0] = (int)(1u << 16);                     // wg_mask=0, data_size=1 (2 bytes)
    g1[1] = (int)((tdim0 & 0xFFFFu) << 16);      // tensor_dim0[15:0] at 63:48
    g1[2] = (int)(((tdim0 >> 16) & 0xFFFFu) | ((tdim1 & 0xFFFFu) << 16));
    g1[3] = (int)(((tdim1 >> 16) & 0xFFFFu) | ((tile_d0 & 0xFFFFu) << 16));
    g1[4] = (int)(tile_d1 & 0xFFFFu);            // tile_dim1 | tile_dim2=0
    g1[5] = (int)(stride0 & 0xFFFFFFFFull);      // tensor_dim0_stride[31:0]
    g1[6] = (int)((stride0 >> 32) & 0xFFFFull);  // stride0[47:32] | stride1.lo=0
    g1[7] = 0;
    v4i z = {0, 0, 0, 0};
#if __clang_major__ >= 23
    v8i z8 = {0, 0, 0, 0, 0, 0, 0, 0};
    __builtin_amdgcn_tensor_load_to_lds(g0, g1, z, z, z8, 0);
#else
    __builtin_amdgcn_tensor_load_to_lds(g0, g1, z, z, 0);
#endif
}

// ---------------- bf16 GEMM: C[M,N] = X[M,K] @ W[N,K]^T (fp32 out) ----------------
// block = 256 threads (8 waves), tile 128x128, K-step 32.
// Double-buffered LDS tiles: TDM prefetch of tile i+1 overlaps compute of tile i;
// wave 0 drains TENSORcnt only to 2 (the in-flight prefetch pair).
__global__ __launch_bounds__(256) void k_gemm_xwt(const __bf16* __restrict__ X,
                                                  const __bf16* __restrict__ W,
                                                  float* __restrict__ C,
                                                  int M, int N, int K) {
    __shared__ __align__(64) __bf16 As[2][128 * 32];
    __shared__ __align__(64) __bf16 Bs[2][128 * 32];
    const int tid = threadIdx.x;
    const int wid = tid >> 5;
    const int lane = tid & 31;
    const int half16 = lane >> 4;
    const int l15 = lane & 15;
    const int m_blk = blockIdx.y * 128;
    const int n_blk = blockIdx.x * 128;
    const int waveM = wid >> 2;    // 0..1
    const int waveN = wid & 3;     // 0..3

    v8f zero = {};
    v8f acc[4][2];
#pragma unroll
    for (int i = 0; i < 4; i++)
#pragma unroll
        for (int j = 0; j < 2; j++) acc[i][j] = zero;

    const unsigned ldsA = (unsigned)(size_t)(&As[0][0]);
    const unsigned ldsB = (unsigned)(size_t)(&Bs[0][0]);

    const int niter = K / 32;
    if (wid == 0) {
        // prime buffer 0
        tdm_load_2d(ldsA, X + (size_t)m_blk * K, 32u, 128u,
                    (unsigned)K, (unsigned)M, (unsigned long long)K);
        tdm_load_2d(ldsB, W + (size_t)n_blk * K, 32u, 128u,
                    (unsigned)K, (unsigned)N, (unsigned long long)K);
    }

    for (int i = 0; i < niter; i++) {
        const int cur = i & 1;
        __syncthreads();   // everyone done reading buf[cur^1] (iter i-1) before overwrite
        if (wid == 0) {
            if (i + 1 < niter) {
                const int ks = (i + 1) * 32;
                tdm_load_2d(ldsA + (unsigned)((cur ^ 1) * 8192),
                            X + (size_t)m_blk * K + ks, 32u, 128u,
                            (unsigned)K, (unsigned)M, (unsigned long long)K);
                tdm_load_2d(ldsB + (unsigned)((cur ^ 1) * 8192),
                            W + (size_t)n_blk * K + ks, 32u, 128u,
                            (unsigned)K, (unsigned)N, (unsigned long long)K);
                __builtin_amdgcn_s_wait_tensorcnt(2);  // buf[cur] landed; prefetch in flight
            } else {
                __builtin_amdgcn_s_wait_tensorcnt(0);
            }
        }
        __syncthreads();   // release buf[cur] to all waves

        const __bf16* Ab = &As[cur][0];
        const __bf16* Bb = &Bs[cur][0];
#pragma unroll
        for (int mi = 0; mi < 4; mi++) {
            // A fragment: lane half 0 -> K {0..7,16..23}; half 1 -> K {8..15,24..31}
            const __bf16* ap = Ab + (waveM * 64 + mi * 16 + l15) * 32 + half16 * 8;
            v8bf alo = *(const v8bf*)ap;
            v8bf ahi = *(const v8bf*)(ap + 16);
            v16bf a;
#pragma unroll
            for (int e = 0; e < 8; e++) { a[e] = alo[e]; a[8 + e] = ahi[e]; }
#pragma unroll
            for (int ni = 0; ni < 2; ni++) {
                // B fragment: lane half 0 -> K {0..7,16..23}; half 1 -> K {8..15,24..31}
                const __bf16* bp = Bb + (waveN * 32 + ni * 16 + l15) * 32 + half16 * 8;
                v8bf blo = *(const v8bf*)bp;
                v8bf bhi = *(const v8bf*)(bp + 16);
                v16bf b;
#pragma unroll
                for (int e = 0; e < 8; e++) { b[e] = blo[e]; b[8 + e] = bhi[e]; }
                acc[mi][ni] = __builtin_amdgcn_wmma_f32_16x16x32_bf16(
                    false, a, false, b, (short)0, acc[mi][ni], false, false);
            }
        }
    }
#pragma unroll
    for (int mi = 0; mi < 4; mi++)
#pragma unroll
        for (int ni = 0; ni < 2; ni++)
#pragma unroll
            for (int r = 0; r < 8; r++) {
                int row = m_blk + waveM * 64 + mi * 16 + half16 * 8 + r;
                int col = n_blk + waveN * 32 + ni * 16 + l15;
                C[(size_t)row * N + col] = acc[mi][ni][r];
            }
}

// ---------------- RMS norm + RoPE for Q -> [B,HQ,S,D] bf16 ----------------
__global__ __launch_bounds__(256) void k_rmsrope_q(const float* __restrict__ Qraw,
                                                   const float* __restrict__ cosb,
                                                   const float* __restrict__ sinb,
                                                   const float* __restrict__ nw,
                                                   __bf16* __restrict__ Qn) {
    const int tid = threadIdx.x, wid = tid >> 5, lane = tid & 31;
    int rid = blockIdx.x * 8 + wid;          // 0..B*HQ*S-1
    int b = rid >> 14;                        // / (HQ*S)
    int rem = rid & 16383;
    int h = rem >> 11;                        // / S
    int s = rem & 2047;
    const float* src = Qraw + ((size_t)(b * CS + s)) * (CHQ * CD) + h * CD;
    float x[8];
    float ss = 0.f;
#pragma unroll
    for (int i = 0; i < 8; i++) { x[i] = src[i * 32 + lane]; ss += x[i] * x[i]; }
    ss += __shfl_xor(ss, 1); ss += __shfl_xor(ss, 2); ss += __shfl_xor(ss, 4);
    ss += __shfl_xor(ss, 8); ss += __shfl_xor(ss, 16);
    float r = rsqrtf(ss * (1.0f / 256.0f) + 1e-6f);
    float n[8];
#pragma unroll
    for (int i = 0; i < 8; i++) {
        int d = i * 32 + lane;
        n[i] = x[i] * r * (1.0f + nw[d]);
    }
    __bf16* dst = Qn + (((size_t)(b * CHQ + h)) * CS + s) * CD;
    const float* cp = cosb + (size_t)s * CD;
    const float* sp = sinb + (size_t)s * CD;
#pragma unroll
    for (int i = 0; i < 8; i++) {
        int d = i * 32 + lane;
        float rot = (i < 4) ? -n[i + 4] : n[i - 4];   // partner d +/- 128 is element i^4
        dst[d] = (__bf16)(n[i] * cp[d] + rot * sp[d]);
    }
}

// ---------------- RMS norm (+RoPE for self) for merged K -> [B,HKV,L,D] bf16 ----------------
__global__ __launch_bounds__(256) void k_rms_k(const float* __restrict__ Kself,
                                               const float* __restrict__ Kcross,
                                               const float* __restrict__ cosb,
                                               const float* __restrict__ sinb,
                                               const float* __restrict__ nw,
                                               __bf16* __restrict__ Ko) {
    const int tid = threadIdx.x, wid = tid >> 5, lane = tid & 31;
    int rid = blockIdx.x * 8 + wid;          // 0..B*HKV*L-1
    int b = rid >> 14;                        // / (HKV*L)
    int rem = rid & 16383;
    int kh = rem >> 12;                       // / L
    int j = rem & 4095;
    bool self = (j < CS);
    const float* src = self
        ? (Kself + ((size_t)(b * CS + j)) * (CHKV * CD) + kh * CD)
        : (Kcross + ((size_t)(b * CENC + (j - CS))) * (CHKV * CD) + kh * CD);
    float x[8];
    float ss = 0.f;
#pragma unroll
    for (int i = 0; i < 8; i++) { x[i] = src[i * 32 + lane]; ss += x[i] * x[i]; }
    ss += __shfl_xor(ss, 1); ss += __shfl_xor(ss, 2); ss += __shfl_xor(ss, 4);
    ss += __shfl_xor(ss, 8); ss += __shfl_xor(ss, 16);
    float r = rsqrtf(ss * (1.0f / 256.0f) + 1e-6f);
    float n[8];
#pragma unroll
    for (int i = 0; i < 8; i++) {
        int d = i * 32 + lane;
        n[i] = x[i] * r * (1.0f + nw[d]);
    }
    __bf16* dst = Ko + (((size_t)(b * CHKV + kh)) * CL + j) * CD;
    if (self) {
        const float* cp = cosb + (size_t)j * CD;
        const float* sp = sinb + (size_t)j * CD;
#pragma unroll
        for (int i = 0; i < 8; i++) {
            int d = i * 32 + lane;
            float rot = (i < 4) ? -n[i + 4] : n[i - 4];
            dst[d] = (__bf16)(n[i] * cp[d] + rot * sp[d]);
        }
    } else {
#pragma unroll
        for (int i = 0; i < 8; i++) dst[i * 32 + lane] = (__bf16)n[i];
    }
}

// ---------------- V pack: merged [B,HKV,L,D] bf16 ----------------
__global__ void k_vpack(const float* __restrict__ Vself, const float* __restrict__ Vcross,
                        __bf16* __restrict__ Vo, int total) {
    int idx = blockIdx.x * blockDim.x + threadIdx.x;
    if (idx >= total) return;
    int d = idx & 255;
    int j = (idx >> 8) & 4095;
    int kh = (idx >> 20) & 3;
    int b = idx >> 22;
    float v = (j < CS)
        ? Vself[((size_t)(b * CS + j)) * (CHKV * CD) + kh * CD + d]
        : Vcross[((size_t)(b * CENC + (j - CS))) * (CHKV * CD) + kh * CD + d];
    Vo[idx] = (__bf16)v;
}

// ---------------- flash attention (softcap + causal-in-self mask) ----------------
// grid = B*HQ*(S/128), block = 256 (8 waves x 16 q-rows), streaming K blocks of 32.
__global__ __launch_bounds__(256) void k_flash(const __bf16* __restrict__ Qn,
                                               const __bf16* __restrict__ Kc,
                                               const __bf16* __restrict__ Vc,
                                               __bf16* __restrict__ Obf) {
    __shared__ __align__(64) __bf16 Vt[CD * 32];        // V^T tile: [d][j]   16KB
    __shared__ __align__(64) __bf16 Ps[8 * 16 * 32];    // P staging per wave  8KB
    const int tid = threadIdx.x, wid = tid >> 5, lane = tid & 31;
    const int half16 = lane >> 4, l15 = lane & 15;
    const int nQB = CS / 128;                 // 16
    const int qb = blockIdx.x % nQB;
    const int bh = blockIdx.x / nQB;
    const int b = bh / CHQ, h = bh % CHQ, kh = h >> 1;  // n_rep = 2

    const __bf16* Qg = Qn + (((size_t)(b * CHQ + h)) * CS + qb * 128) * CD;
    const __bf16* Kg = Kc + ((size_t)(b * CHKV + kh)) * CL * CD;
    const __bf16* Vg = Vc + ((size_t)(b * CHKV + kh)) * CL * CD;

    const int m0 = wid * 16;
    // Q A-fragments for this wave's 16 rows, held in registers (8 d-chunks of 32)
    v16bf qf[8];
#pragma unroll
    for (int dc = 0; dc < 8; dc++) {
        const __bf16* ap = Qg + (size_t)(m0 + l15) * CD + dc * 32 + half16 * 8;
        v8bf alo = *(const v8bf*)ap;
        v8bf ahi = *(const v8bf*)(ap + 16);
#pragma unroll
        for (int e = 0; e < 8; e++) { qf[dc][e] = alo[e]; qf[dc][8 + e] = ahi[e]; }
    }

    v8f zero = {};
    v8f o[16];
#pragma unroll
    for (int t = 0; t < 16; t++) o[t] = zero;
    float mrow[8], lrow[8];
#pragma unroll
    for (int r = 0; r < 8; r++) { mrow[r] = -1e30f; lrow[r] = 0.0f; }

    const int irow_base = qb * 128 + m0 + half16 * 8;
    const int qmax = qb * 128 + 127;

    for (int jb = 0; jb < CL; jb += 32) {
        if (jb < CS && jb > qmax) continue;   // fully-masked self block (uniform)

        // speculative prefetch of next K block (per-lane row starts)
        if (jb + 32 < CL) {
            __builtin_prefetch(Kg + (size_t)(jb + 32 + lane) * CD, 0, 0);
        }

        __syncthreads();
        // stage V^T tile
        for (int idx = tid; idx < 32 * CD; idx += 256) {
            int j = idx >> 8, d = idx & 255;
            Vt[d * 32 + j] = Vg[(size_t)(jb + j) * CD + d];
        }
        __syncthreads();

        // scores: two 16x16 tiles covering 32 keys
        v8f c0 = zero, c1 = zero;
#pragma unroll
        for (int dc = 0; dc < 8; dc++) {
            const __bf16* bp0 = Kg + (size_t)(jb + l15) * CD + dc * 32 + half16 * 8;
            const __bf16* bp1 = Kg + (size_t)(jb + 16 + l15) * CD + dc * 32 + half16 * 8;
            v8bf b0lo = *(const v8bf*)bp0, b0hi = *(const v8bf*)(bp0 + 16);
            v8bf b1lo = *(const v8bf*)bp1, b1hi = *(const v8bf*)(bp1 + 16);
            v16bf b0, b1;
#pragma unroll
            for (int e = 0; e < 8; e++) { b0[e] = b0lo[e]; b0[8 + e] = b0hi[e]; b1[e] = b1lo[e]; b1[8 + e] = b1hi[e]; }
            c0 = __builtin_amdgcn_wmma_f32_16x16x32_bf16(false, qf[dc], false, b0,
                                                         (short)0, c0, false, false);
            c1 = __builtin_amdgcn_wmma_f32_16x16x32_bf16(false, qf[dc], false, b1,
                                                         (short)0, c1, false, false);
        }

        // softcap + mask + online softmax (rows live in 16-lane halves)
        float scl[8];
#pragma unroll
        for (int r = 0; r < 8; r++) {
            int i = irow_base + r;
            int j0 = jb + l15;
            int j1 = jb + 16 + l15;
            float s0 = SOFTCAP * tanhf(c0[r] * (SCALING / SOFTCAP));
            float s1 = SOFTCAP * tanhf(c1[r] * (SCALING / SOFTCAP));
            if (j0 < CS && j0 > i) s0 = NEGINF;
            if (j1 < CS && j1 > i) s1 = NEGINF;
            float rm = fmaxf(s0, s1);
            rm = fmaxf(rm, __shfl_xor(rm, 1));
            rm = fmaxf(rm, __shfl_xor(rm, 2));
            rm = fmaxf(rm, __shfl_xor(rm, 4));
            rm = fmaxf(rm, __shfl_xor(rm, 8));
            float mnew = fmaxf(mrow[r], rm);
            float sc = __expf(mrow[r] - mnew);
            float p0 = __expf(s0 - mnew);
            float p1 = __expf(s1 - mnew);
            float ps = p0 + p1;
            ps += __shfl_xor(ps, 1); ps += __shfl_xor(ps, 2);
            ps += __shfl_xor(ps, 4); ps += __shfl_xor(ps, 8);
            lrow[r] = lrow[r] * sc + ps;
            mrow[r] = mnew;
            scl[r] = sc;
            // stage P (C-layout -> row-major 16x32) for A-fragment reload
            Ps[wid * 512 + (half16 * 8 + r) * 32 + l15] = (__bf16)p0;
            Ps[wid * 512 + (half16 * 8 + r) * 32 + 16 + l15] = (__bf16)p1;
        }
#pragma unroll
        for (int t = 0; t < 16; t++)
#pragma unroll
            for (int r = 0; r < 8; r++) o[t][r] *= scl[r];

        asm volatile("s_wait_dscnt 0" ::: "memory");   // P writes -> P reads (same wave)

        v16bf pf;
        {
            const __bf16* pp = &Ps[wid * 512 + l15 * 32 + half16 * 8];
            v8bf plo = *(const v8bf*)pp;
            v8bf phi = *(const v8bf*)(pp + 16);
#pragma unroll
            for (int e = 0; e < 8; e++) { pf[e] = plo[e]; pf[8 + e] = phi[e]; }
        }
        // O += P @ V  (16 d-tiles of 16)
#pragma unroll
        for (int t = 0; t < 16; t++) {
            const __bf16* vp = &Vt[(t * 16 + l15) * 32 + half16 * 8];
            v8bf vlo = *(const v8bf*)vp;
            v8bf vhi = *(const v8bf*)(vp + 16);
            v16bf vb;
#pragma unroll
            for (int e = 0; e < 8; e++) { vb[e] = vlo[e]; vb[8 + e] = vhi[e]; }
            o[t] = __builtin_amdgcn_wmma_f32_16x16x32_bf16(false, pf, false, vb,
                                                           (short)0, o[t], false, false);
        }
    }

    // normalize + write [B*S, HQ*D] bf16
#pragma unroll
    for (int r = 0; r < 8; r++) {
        float inv = 1.0f / lrow[r];
        int row = b * CS + qb * 128 + m0 + half16 * 8 + r;
#pragma unroll
        for (int t = 0; t < 16; t++) {
            Obf[(size_t)row * (CHQ * CD) + h * CD + t * 16 + l15] = (__bf16)(o[t][r] * inv);
        }
    }
}

// ---------------- launch ----------------
extern "C" void kernel_launch(void* const* d_in, const int* in_sizes, int n_in,
                              void* d_out, int out_size, void* d_ws, size_t ws_size,
                              hipStream_t stream) {
    (void)in_sizes; (void)n_in; (void)out_size; (void)ws_size;
    const float* hidden = (const float*)d_in[0];
    const float* enc    = (const float*)d_in[1];
    const float* cosb   = (const float*)d_in[2];
    const float* sinb   = (const float*)d_in[3];
    // d_in[4] merged_attention_mask: computed analytically, not read
    const float* Wq = (const float*)d_in[5];
    const float* Wk = (const float*)d_in[6];
    const float* Wv = (const float*)d_in[7];
    const float* Wo = (const float*)d_in[8];
    const float* qw = (const float*)d_in[9];
    const float* kw = (const float*)d_in[10];

    char* ws = (char*)d_ws;
    __bf16* hid_bf = (__bf16*)(ws + OFF_HIDBF);
    __bf16* enc_bf = (__bf16*)(ws + OFF_ENCBF);
    __bf16* wq_bf  = (__bf16*)(ws + OFF_WQBF);
    __bf16* wk_bf  = (__bf16*)(ws + OFF_WKBF);
    __bf16* wv_bf  = (__bf16*)(ws + OFF_WVBF);
    __bf16* wo_bf  = (__bf16*)(ws + OFF_WOBF);
    float*  qraw   = (float*)(ws + OFF_QRAW);
    float*  kself  = (float*)(ws + OFF_KSELF);
    float*  vself  = (float*)(ws + OFF_VSELF);
    float*  kcross = (float*)(ws + OFF_KCROSS);
    float*  vcross = (float*)(ws + OFF_VCROSS);
    __bf16* qn     = (__bf16*)(ws + OFF_QN);
    __bf16* kbf    = (__bf16*)(ws + OFF_KBF);
    __bf16* vbf    = (__bf16*)(ws + OFF_VBF);
    __bf16* attnbf = (__bf16*)(ws + OFF_ATTN);

    const int nHid = CB * CS * CHID;           // 8388608
    const int nWq  = CHQ * CD * CHID;          // 4194304
    const int nWk  = CHKV * CD * CHID;         // 2097152

    k_f2bf<<<(nHid / 4 + 255) / 256, 256, 0, stream>>>(hidden, hid_bf, nHid);
    k_f2bf<<<(nHid / 4 + 255) / 256, 256, 0, stream>>>(enc, enc_bf, nHid);
    k_f2bf<<<(nWq / 4 + 255) / 256, 256, 0, stream>>>(Wq, wq_bf, nWq);
    k_f2bf<<<(nWk / 4 + 255) / 256, 256, 0, stream>>>(Wk, wk_bf, nWk);
    k_f2bf<<<(nWk / 4 + 255) / 256, 256, 0, stream>>>(Wv, wv_bf, nWk);
    k_f2bf<<<(nWq / 4 + 255) / 256, 256, 0, stream>>>(Wo, wo_bf, nWq);

    const int M = CB * CS;                     // 4096
    // projections
    k_gemm_xwt<<<dim3(2048 / 128, M / 128), 256, 0, stream>>>(hid_bf, wq_bf, qraw, M, 2048, CHID);
    k_gemm_xwt<<<dim3(1024 / 128, M / 128), 256, 0, stream>>>(hid_bf, wk_bf, kself, M, 1024, CHID);
    k_gemm_xwt<<<dim3(1024 / 128, M / 128), 256, 0, stream>>>(hid_bf, wv_bf, vself, M, 1024, CHID);
    k_gemm_xwt<<<dim3(1024 / 128, M / 128), 256, 0, stream>>>(enc_bf, wk_bf, kcross, M, 1024, CHID);
    k_gemm_xwt<<<dim3(1024 / 128, M / 128), 256, 0, stream>>>(enc_bf, wv_bf, vcross, M, 1024, CHID);

    // norm / rope / pack
    k_rmsrope_q<<<(CB * CHQ * CS) / 8, 256, 0, stream>>>(qraw, cosb, sinb, qw, qn);
    k_rms_k<<<(CB * CHKV * CL) / 8, 256, 0, stream>>>(kself, kcross, cosb, sinb, kw, kbf);
    k_vpack<<<(CB * CHKV * CL * CD + 255) / 256, 256, 0, stream>>>(vself, vcross, vbf,
                                                                   CB * CHKV * CL * CD);
    // attention
    k_flash<<<CB * CHQ * (CS / 128), 256, 0, stream>>>(qn, kbf, vbf, attnbf);

    // output projection -> d_out (fp32)
    k_gemm_xwt<<<dim3(2048 / 128, M / 128), 256, 0, stream>>>(attnbf, wo_bf, (float*)d_out,
                                                              M, 2048, CHID);
}
